// AttentionDecoderRNN_88338887344394
// MI455X (gfx1250) — matrix operand order repaired and checked by reading, not compiled, not run on hardware
//
#include <hip/hip_runtime.h>
#include <hip/hip_bf16.h>
#include <math.h>

typedef __attribute__((ext_vector_type(16))) __bf16 bf16x16;
typedef __attribute__((ext_vector_type(8)))  float  f32x8;

union FragBF {
  bf16x16 v;
  uint4   u[2];
};

static constexpr int B_   = 64;
static constexpr int P_   = 49;
static constexpr int ENC_ = 2048;
static constexpr int EMB_ = 512;
static constexpr int HID_ = 512;
static constexpr int ATT_ = 512;
static constexpr int V_   = 30000;
static constexpr int T_   = 30;
static constexpr int CAP_ = 31;
static constexpr int KX_  = EMB_ + ENC_ + HID_;   // 3072 = [emb | context | h]
static constexpr int G_   = 4 * HID_;             // 2048 gate width

#define WMMA_BF16(A, Bf, Acc) \
  __builtin_amdgcn_wmma_f32_16x16x32_bf16(false, (A).v, false, (Bf).v, (short)0, (Acc), false, false)

// ---------------------------------------------------------------------------
// bf16 WMMA GEMM:  C[M,N] = A[M,K] (row-major, lda) @ Bpacked[K,Npad] + bias
// Bpacked: zero-padded to Npad (multiple of 128); per (kblock, n) a contiguous
// 32-element K-panel: Bp[ ((k>>5)*Npad + n)*32 + (k&31) ].
// Block = 256 threads = 8 wave32; block tile 64(M) x 128(N);
// wave (wm = wid&3, wn = wid>>2) owns rows [m0,m0+16) x cols [nbase,nbase+64).
// K-loop unrolled x2 with ping-pong fragment sets: no register rotation moves,
// each WMMA group waits only on its own fragment loads.  Requires K % 64 == 0.
// ---------------------------------------------------------------------------
__global__ void __launch_bounds__(256)
gemm_bf16_wmma(const __bf16* __restrict__ A, int lda,
               const __bf16* __restrict__ Bp,
               const float* __restrict__ bias,
               float* __restrict__ C, long ldc,
               int M, int N, int Npad, int K)
{
  const int tid  = threadIdx.x;
  const int lane = tid & 31;
  const int wid  = tid >> 5;
  const int wm   = wid & 3;
  const int wn   = wid >> 2;

  const int m0    = blockIdx.y * 64 + wm * 16;
  const int nbase = blockIdx.x * 128 + wn * 64;
  if (m0 >= M) return;

  const int mrow = lane & 15;        // A row within tile / B,C column
  const int half = lane >> 4;        // selects K sub-range within fragment

  const __bf16* arow   = A + (long)(m0 + mrow) * lda + half * 16;
  const __bf16* bbase  = Bp + (long)(nbase + mrow) * 32 + half * 16;
  const long    bkstep = (long)Npad * 32;          // elements per k-block of 32

  f32x8 acc0 = {}, acc1 = {}, acc2 = {}, acc3 = {};

  FragBF a0, b00, b01, b02, b03;     // fragment set 0
  FragBF a1, b10, b11, b12, b13;     // fragment set 1

  auto ldA = [&](FragBF& f, int k) {
    f.u[0] = *reinterpret_cast<const uint4*>(arow + k);
    f.u[1] = *reinterpret_cast<const uint4*>(arow + k + 8);
  };
  auto ldB = [&](FragBF& f0, FragBF& f1, FragBF& f2, FragBF& f3, int k) {
    const __bf16* bp = bbase + (long)(k >> 5) * bkstep;
    f0.u[0] = *reinterpret_cast<const uint4*>(bp);
    f0.u[1] = *reinterpret_cast<const uint4*>(bp + 8);
    f1.u[0] = *reinterpret_cast<const uint4*>(bp + 512);
    f1.u[1] = *reinterpret_cast<const uint4*>(bp + 520);
    f2.u[0] = *reinterpret_cast<const uint4*>(bp + 1024);
    f2.u[1] = *reinterpret_cast<const uint4*>(bp + 1032);
    f3.u[0] = *reinterpret_cast<const uint4*>(bp + 1536);
    f3.u[1] = *reinterpret_cast<const uint4*>(bp + 1544);
  };

  // prologue: fragment set 0 <- k-block 0
  ldA(a0, 0);
  ldB(b00, b01, b02, b03, 0);

  for (int k0 = 0; k0 < K; k0 += 64) {
    // stage set 1 (k0+32) while computing on set 0
    ldA(a1, k0 + 32);
    ldB(b10, b11, b12, b13, k0 + 32);

    acc0 = WMMA_BF16(a0, b00, acc0);
    acc1 = WMMA_BF16(a0, b01, acc1);
    acc2 = WMMA_BF16(a0, b02, acc2);
    acc3 = WMMA_BF16(a0, b03, acc3);

    // stage set 0 (k0+64) while computing on set 1
    if (k0 + 64 < K) {                              // wave-uniform scalar branch
      ldA(a0, k0 + 64);
      ldB(b00, b01, b02, b03, k0 + 64);
    }

    acc0 = WMMA_BF16(a1, b10, acc0);
    acc1 = WMMA_BF16(a1, b11, acc1);
    acc2 = WMMA_BF16(a1, b12, acc2);
    acc3 = WMMA_BF16(a1, b13, acc3);
  }

  // Store (edge-guarded per column): VGPR r -> row m0 + half*8 + r
#pragma unroll
  for (int j = 0; j < 4; ++j) {
    const int col = nbase + j * 16 + mrow;
    if (col < N) {
      const f32x8& a = (j == 0) ? acc0 : (j == 1) ? acc1 : (j == 2) ? acc2 : acc3;
      const float bv = bias ? bias[col] : 0.0f;
      float* crow = C + (long)(m0 + half * 8) * ldc + col;
#pragma unroll
      for (int r = 0; r < 8; ++r)
        crow[(long)r * ldc] = a[r] + bv;
    }
  }
}

// ---------------------------------------------------------------------------
// Pack a K x N f32 matrix (optionally transposed source) into zero-padded
// bf16 K-panels of width Npad.
// ---------------------------------------------------------------------------
__global__ void pack_b_bf16(const float* __restrict__ src, __bf16* __restrict__ dst,
                            int Ksrc, int N, int Npad, int ld, int trans, int kOff)
{
  long i = (long)blockIdx.x * 256 + threadIdx.x;
  if (i >= (long)Ksrc * Npad) return;
  const int k = (int)(i / Npad);
  const int n = (int)(i % Npad);
  float v = 0.0f;
  if (n < N) v = trans ? src[(long)n * ld + k] : src[(long)k * ld + n];
  const int kg = k + kOff;
  dst[((long)(kg >> 5) * Npad + n) * 32 + (kg & 31)] = (__bf16)v;
}

__global__ void f32_to_bf16_kernel(const float* __restrict__ src, __bf16* __restrict__ dst, long n)
{
  long i = (long)blockIdx.x * 256 + threadIdx.x;
  if (i < n) dst[i] = (__bf16)src[i];
}

// emb_bf[b,t,e] = bf16(embed_W[captions[b,t], e]) for t in [0,30)
__global__ void embed_gather(const float* __restrict__ embW, const int* __restrict__ caps,
                             __bf16* __restrict__ embBf)
{
  long i = (long)blockIdx.x * 256 + threadIdx.x;
  if (i >= (long)B_ * T_ * EMB_) return;
  const int e  = (int)(i & (EMB_ - 1));
  const int bt = (int)(i >> 9);
  const int t  = bt % T_;
  const int b  = bt / T_;
  const int tok = caps[b * CAP_ + t];
  embBf[i] = (__bf16)embW[(long)tok * EMB_ + e];
}

// zero h, c (f32) and the h-slice of xh (bf16); combine gate biases
__global__ void init_state(float* __restrict__ h, float* __restrict__ c,
                           __bf16* __restrict__ xh,
                           const float* __restrict__ bih, const float* __restrict__ bhh,
                           float* __restrict__ biasg)
{
  const int i = blockIdx.x * 256 + threadIdx.x;
  if (i < B_ * HID_) {
    h[i] = 0.0f;
    c[i] = 0.0f;
    xh[(long)(i / HID_) * KX_ + (EMB_ + ENC_) + (i % HID_)] = (__bf16)0.0f;
  }
  if (i < G_) biasg[i] = bih[i] + bhh[i];
}

// ---------------------------------------------------------------------------
// Fused attention step (one block per batch row b):
//   att2 = h @ Wdec + bdec ; e = relu(att1 + att2) @ Wfull + bfull
//   alpha = softmax_P(e)   ; context = alpha @ enc
// Writes alpha to d_out, context + emb_t (bf16) into xh staging row.
// ---------------------------------------------------------------------------
__global__ void __launch_bounds__(256)
attention_step(const float* __restrict__ att1, const float* __restrict__ enc,
               const float* __restrict__ h,
               const float* __restrict__ Wdec, const float* __restrict__ bdec,
               const float* __restrict__ Wfull, const float* __restrict__ bfull,
               const __bf16* __restrict__ embBf,
               __bf16* __restrict__ xh, float* __restrict__ alphaOut, int t)
{
  __shared__ float hsh[HID_];
  __shared__ float a2sh[ATT_];
  __shared__ float esh[64];

  const int b    = blockIdx.x;
  const int tid  = threadIdx.x;
  const int lane = tid & 31;
  const int wid  = tid >> 5;

  for (int i = tid; i < HID_; i += 256) hsh[i] = h[b * HID_ + i];
  __syncthreads();

  // att2: each thread computes 2 columns
  for (int ci = tid; ci < ATT_; ci += 256) {
    float acc = bdec[ci];
    for (int k = 0; k < HID_; ++k)
      acc += hsh[k] * Wdec[(long)k * ATT_ + ci];
    a2sh[ci] = acc;
  }
  __syncthreads();

  // e[p]: one wave per p, strided over p
  for (int p = wid; p < P_; p += 8) {
    const float* row = att1 + ((long)b * P_ + p) * ATT_;
    float s = 0.0f;
    for (int a = lane; a < ATT_; a += 32) {
      const float r = row[a] + a2sh[a];
      s += (r > 0.0f ? r : 0.0f) * Wfull[a];
    }
    for (int off = 16; off; off >>= 1) s += __shfl_down(s, off, 32);
    if (lane == 0) esh[p] = s + bfull[0];
  }
  __syncthreads();

  // softmax over P=49 in wave 0
  if (wid == 0) {
    float v0 = (lane < P_)      ? esh[lane]      : -3.0e38f;
    float v1 = (lane + 32 < P_) ? esh[lane + 32] : -3.0e38f;
    float m = fmaxf(v0, v1);
    for (int off = 16; off; off >>= 1) m = fmaxf(m, __shfl_xor(m, off, 32));
    float x0 = (lane < P_)      ? __expf(v0 - m) : 0.0f;
    float x1 = (lane + 32 < P_) ? __expf(v1 - m) : 0.0f;
    float s = x0 + x1;
    for (int off = 16; off; off >>= 1) s += __shfl_xor(s, off, 32);
    const float inv = 1.0f / s;
    if (lane < P_) {
      const float a0 = x0 * inv;
      esh[lane] = a0;
      alphaOut[((long)b * T_ + t) * P_ + lane] = a0;
    }
    if (lane + 32 < P_) {
      const float a1 = x1 * inv;
      esh[lane + 32] = a1;
      alphaOut[((long)b * T_ + t) * P_ + lane + 32] = a1;
    }
  }
  __syncthreads();

  // context -> xh[:, EMB:EMB+ENC] (bf16)
  __bf16* xrow = xh + (long)b * KX_;
  for (int e = tid; e < ENC_; e += 256) {
    float acc = 0.0f;
    const float* ecol = enc + ((long)b * P_) * ENC_ + e;
    for (int p = 0; p < P_; ++p)
      acc += esh[p] * ecol[(long)p * ENC_];
    xrow[EMB_ + e] = (__bf16)acc;
  }
  // emb_t -> xh[:, 0:EMB]
  const __bf16* erow = embBf + ((long)b * T_ + t) * EMB_;
  for (int i = tid; i < EMB_; i += 256) xrow[i] = erow[i];
}

// LSTM pointwise: c,h update + write h (bf16) back into xh h-slice
__global__ void __launch_bounds__(256)
lstm_step(const float* __restrict__ gates, float* __restrict__ h, float* __restrict__ c,
          __bf16* __restrict__ xh)
{
  const int b = blockIdx.x;
  const float* g = gates + (long)b * G_;
  for (int u = threadIdx.x; u < HID_; u += 256) {
    const float i_ = g[u];
    const float f_ = g[HID_ + u];
    const float g_ = g[2 * HID_ + u];
    const float o_ = g[3 * HID_ + u];
    const float si = 1.0f / (1.0f + __expf(-i_));
    const float sf = 1.0f / (1.0f + __expf(-f_));
    const float so = 1.0f / (1.0f + __expf(-o_));
    const float cn = sf * c[(long)b * HID_ + u] + si * tanhf(g_);
    const float hn = so * tanhf(cn);
    c[(long)b * HID_ + u] = cn;
    h[(long)b * HID_ + u] = hn;
    xh[(long)b * KX_ + (EMB_ + ENC_) + u] = (__bf16)hn;
  }
}

// ---------------------------------------------------------------------------
extern "C" void kernel_launch(void* const* d_in, const int* in_sizes, int n_in,
                              void* d_out, int out_size, void* d_ws, size_t ws_size,
                              hipStream_t stream)
{
  (void)in_sizes; (void)n_in; (void)out_size; (void)ws_size;
  const float* enc   = (const float*)d_in[0];
  const int*   caps  = (const int*)  d_in[1];
  const float* embW  = (const float*)d_in[3];
  const float* Wenc  = (const float*)d_in[4];
  const float* benc  = (const float*)d_in[5];
  const float* Wdec  = (const float*)d_in[6];
  const float* bdec  = (const float*)d_in[7];
  const float* Wfull = (const float*)d_in[8];
  const float* bfull = (const float*)d_in[9];
  const float* Wih   = (const float*)d_in[10];
  const float* Whh   = (const float*)d_in[11];
  const float* bih   = (const float*)d_in[12];
  const float* bhh   = (const float*)d_in[13];
  const float* Wlin  = (const float*)d_in[14];
  const float* blin  = (const float*)d_in[15];

  constexpr int VPAD = ((V_ + 127) / 128) * 128;   // 30080

  char* ws = (char*)d_ws;
  size_t off = 0;
  auto alloc = [&](size_t bytes) -> char* {
    char* p = ws + off;
    off = (off + bytes + 255) & ~(size_t)255;
    return p;
  };
  __bf16* wencP = (__bf16*)alloc((size_t)ENC_ * ATT_ * 2);
  __bf16* wgP   = (__bf16*)alloc((size_t)KX_ * G_ * 2);
  __bf16* wlinP = (__bf16*)alloc((size_t)HID_ * VPAD * 2);
  __bf16* encBf = (__bf16*)alloc((size_t)B_ * P_ * ENC_ * 2);
  __bf16* embBf = (__bf16*)alloc((size_t)B_ * T_ * EMB_ * 2);
  float*  att1  = (float*)alloc((size_t)B_ * P_ * ATT_ * 4);
  float*  hbuf  = (float*)alloc((size_t)B_ * HID_ * 4);
  float*  cbuf  = (float*)alloc((size_t)B_ * HID_ * 4);
  __bf16* xh    = (__bf16*)alloc((size_t)B_ * KX_ * 2);
  float*  gates = (float*)alloc((size_t)B_ * G_ * 4);
  float*  biasg = (float*)alloc((size_t)G_ * 4);

  float* outPreds = (float*)d_out;
  float* outAlpha = outPreds + (size_t)B_ * T_ * V_;

  auto blocksFor = [](long n) { return (unsigned)((n + 255) / 256); };

  // --- one-time (per launch) setup: bf16 conversion + packing -------------
  pack_b_bf16<<<blocksFor((long)ENC_ * ATT_), 256, 0, stream>>>(Wenc, wencP, ENC_, ATT_, ATT_, ATT_, 0, 0);
  pack_b_bf16<<<blocksFor((long)(EMB_ + ENC_) * G_), 256, 0, stream>>>(Wih, wgP, EMB_ + ENC_, G_, G_, EMB_ + ENC_, 1, 0);
  pack_b_bf16<<<blocksFor((long)HID_ * G_), 256, 0, stream>>>(Whh, wgP, HID_, G_, G_, HID_, 1, EMB_ + ENC_);
  pack_b_bf16<<<blocksFor((long)HID_ * VPAD), 256, 0, stream>>>(Wlin, wlinP, HID_, V_, VPAD, V_, 0, 0);
  f32_to_bf16_kernel<<<blocksFor((long)B_ * P_ * ENC_), 256, 0, stream>>>(enc, encBf, (long)B_ * P_ * ENC_);
  embed_gather<<<blocksFor((long)B_ * T_ * EMB_), 256, 0, stream>>>(embW, caps, embBf);
  init_state<<<blocksFor((long)B_ * HID_), 256, 0, stream>>>(hbuf, cbuf, xh, bih, bhh, biasg);

  auto gemm = [&](const __bf16* A, int lda, const __bf16* Bp, const float* bias,
                  float* C, long ldc, int M, int N, int Npad, int K) {
    dim3 grid(Npad / 128, (M + 63) / 64);
    gemm_bf16_wmma<<<grid, 256, 0, stream>>>(A, lda, Bp, bias, C, ldc, M, N, Npad, K);
  };

  // att1 = encoder_out @ Wenc + benc : (3136 x 512), K=2048
  gemm(encBf, ENC_, wencP, benc, att1, ATT_, B_ * P_, ATT_, ATT_, ENC_);

  // --- 30 sequential decoder steps ----------------------------------------
  for (int t = 0; t < T_; ++t) {
    attention_step<<<B_, 256, 0, stream>>>(att1, enc, hbuf, Wdec, bdec, Wfull, bfull,
                                           embBf, xh, outAlpha, t);
    // gates = [emb|ctx|h] @ [Wih^T ; Whh^T] + (b_ih + b_hh) : (64 x 2048), K=3072
    gemm(xh, KX_, wgP, biasg, gates, G_, B_, G_, G_, KX_);
    lstm_step<<<B_, 256, 0, stream>>>(gates, hbuf, cbuf, xh);
    // preds = h_new @ Wlin + blin : (64 x 30000), K=512, written straight to d_out
    gemm(xh + (EMB_ + ENC_), KX_, wlinP, blin, outPreds + (size_t)t * V_,
         (long)T_ * V_, B_, V_, VPAD, HID_);
  }
}
